// VisionTransformer_5574867550986
// MI455X (gfx1250) — matrix operand order repaired and checked by reading, not compiled, not run on hardware
//
#include <hip/hip_runtime.h>
#include <math.h>

#define DEV __device__ __forceinline__

typedef __attribute__((ext_vector_type(4)))  unsigned int u32x4;
typedef __attribute__((ext_vector_type(8)))  float        v8f;
typedef __attribute__((ext_vector_type(16))) __bf16       v16bf;
typedef unsigned short ushort_t;

union FragU { v16bf v; u32x4 u[2]; };
union Pack8 { u32x4 u; unsigned short s[8]; };

DEV unsigned short f2bf(float x) {
    union { float f; unsigned int u; } c; c.f = x;
    unsigned int r = (c.u + 0x7FFFu + ((c.u >> 16) & 1u)) >> 16;
    return (unsigned short)r;
}

static constexpr int BATCH  = 64;
static constexpr int NTOK   = 197;
static constexpr int DMODEL = 768;
static constexpr int NHEAD  = 12;
static constexpr int HDIM   = 64;
static constexpr int DMLP   = 3072;
static constexpr int NDEPTH = 12;
static constexpr int NPATCH = 196;
static constexpr int MROWS  = BATCH * NTOK;    // 12608
static constexpr int MPATCH = BATCH * NPATCH;  // 12544
static constexpr int NCLS   = 1000;
static constexpr float ATTN_SCALE = 0.125f;    // 64^-0.5
static constexpr float LN_EPS = 1e-5f;

// ---------------- WMMA fragment loaders (16-bit, 16x32 A / 32x16 B) ----------
// A-matrix 16x32 bf16 (ISA 7.12.2): lane (m = L&15, half = L>>4);
//   elements 0..7  -> K = 8*half + e        (two b128 loads from row-major)
//   elements 8..15 -> K = 16 + 8*half + (e-8)
DEV v16bf frag_a(const ushort_t* p, int ld) {
    int lane = threadIdx.x & 31;
    const ushort_t* r = p + (lane & 15) * ld + 8 * (lane >> 4);
    FragU f;
    f.u[0] = *(const u32x4*)(r);
    f.u[1] = *(const u32x4*)(r + 16);
    return f.v;
}
// B-matrix 32x16 bf16 from K-major LDS: lane n = L&15, half = L>>4;
// elements 0..7 -> K = 8*half + e, elements 8..15 -> K = 16 + 8*half + (e-8).
DEV v16bf frag_b(const ushort_t* p, int ld) {
    int lane = threadIdx.x & 31;
    const ushort_t* r = p + (lane & 15) + 8 * (lane >> 4) * ld;
    union { v16bf v; ushort_t s[16]; } f;
    #pragma unroll
    for (int e = 0; e < 8; ++e) {
        f.s[e] = r[e * ld];
        f.s[8 + e] = r[(16 + e) * ld];
    }
    return f.v;
}

// ---------------- elementwise kernels ---------------------------------------
__global__ void cvt_kernel(const float* __restrict__ in, ushort_t* __restrict__ out, long n) {
    long i = (long)blockIdx.x * 256 + threadIdx.x;
    if (i < n) out[i] = f2bf(in[i]);
}

__global__ void im2col_kernel(const float* __restrict__ x, ushort_t* __restrict__ Xp) {
    long idx = (long)blockIdx.x * 256 + threadIdx.x;
    if (idx >= (long)MPATCH * DMODEL) return;
    int k = (int)(idx % DMODEL);
    long bp = idx / DMODEL;
    int p = (int)(bp % NPATCH);
    int b = (int)(bp / NPATCH);
    int c = k >> 8, ij = k & 255, i = ij >> 4, j = ij & 15;
    int py = p / 14, px = p % 14;
    float v = x[(((size_t)b * 3 + c) * 224 + py * 16 + i) * 224 + px * 16 + j];
    Xp[idx] = f2bf(v);
}

__global__ void assemble_kernel(const float* __restrict__ P32, const float* __restrict__ cls,
                                const float* __restrict__ pos, float* __restrict__ h) {
    long idx = (long)blockIdx.x * 256 + threadIdx.x;
    if (idx >= (long)MROWS * DMODEL) return;
    int d = (int)(idx % DMODEL);
    long bt = idx / DMODEL;
    int t = (int)(bt % NTOK);
    int b = (int)(bt / NTOK);
    float v = (t == 0) ? cls[d] : P32[((size_t)b * NPATCH + (t - 1)) * DMODEL + d];
    h[idx] = v + pos[(size_t)t * DMODEL + d];
}

// ---------------- LayerNorm: f32 in -> bf16 out ------------------------------
__global__ __launch_bounds__(256)
void ln_kernel(const float* __restrict__ in, long inRowStride,
               const float* __restrict__ sc, const float* __restrict__ bi,
               ushort_t* __restrict__ out, int rows) {
    __shared__ float red[256], red2[256];
    int r = blockIdx.x;
    if (r >= rows) return;
    const float* x = in + (size_t)r * inRowStride;
    float s = 0.f, s2 = 0.f;
    for (int i = threadIdx.x; i < DMODEL; i += 256) { float v = x[i]; s += v; s2 += v * v; }
    red[threadIdx.x] = s; red2[threadIdx.x] = s2;
    __syncthreads();
    for (int off = 128; off > 0; off >>= 1) {
        if (threadIdx.x < off) {
            red[threadIdx.x]  += red[threadIdx.x + off];
            red2[threadIdx.x] += red2[threadIdx.x + off];
        }
        __syncthreads();
    }
    float mean = red[0] * (1.0f / DMODEL);
    float var  = red2[0] * (1.0f / DMODEL) - mean * mean;
    float rs = rsqrtf(var + LN_EPS);
    for (int i = threadIdx.x; i < DMODEL; i += 256)
        out[(size_t)r * DMODEL + i] = f2bf((x[i] - mean) * rs * sc[i] + bi[i]);
}

// ---------------- tiled WMMA GEMM: C[M,N] = A[M,K] * W[N,K]^T + bias ---------
// 256 threads = 8 waves, block tile 128x64, per-wave 32x32 (2x2 WMMA tiles).
template<bool GELU, bool RESID, bool O32, bool O16>
__global__ __launch_bounds__(256)
void gemm_bt_kernel(const ushort_t* __restrict__ A, int lda,
                    const ushort_t* __restrict__ W, int ldw,
                    const float* __restrict__ bias,
                    const float* __restrict__ resid,
                    float* __restrict__ out32,
                    ushort_t* __restrict__ out16,
                    int M, int Nout, int K) {
    __shared__ ushort_t As[128 * 32];  // row-major [128][32]
    __shared__ ushort_t Bs[32 * 64];   // K-major  [32][64] (transposed W tile)
    const int tid  = threadIdx.x;
    const int lane = tid & 31;
    const int wave = tid >> 5;
    const int wm = wave & 3, wn = wave >> 2;
    const int rowBase = blockIdx.y * 128;
    const int colBase = blockIdx.x * 64;

    v8f acc[2][2];
    acc[0][0] = (v8f)0.f; acc[0][1] = (v8f)0.f;
    acc[1][0] = (v8f)0.f; acc[1][1] = (v8f)0.f;

    for (int kt = 0; kt < K; kt += 32) {
        // stage A tile: 128x32 bf16, 512 x 16B chunks
        #pragma unroll
        for (int i = 0; i < 2; ++i) {
            int c = tid + i * 256;
            int row = c >> 2, cc = c & 3;
            u32x4 val = (u32x4)0u;
            int gr = rowBase + row;
            if (gr < M) val = *(const u32x4*)(A + (size_t)gr * lda + kt + cc * 8);
            *(u32x4*)(&As[row * 32 + cc * 8]) = val;
        }
        // stage W tile transposed into K-major Bs
        {
            int n = tid >> 2, cc = tid & 3;
            Pack8 pk; pk.u = (u32x4)0u;
            int gc = colBase + n;
            if (gc < Nout) pk.u = *(const u32x4*)(W + (size_t)gc * ldw + kt + cc * 8);
            #pragma unroll
            for (int e = 0; e < 8; ++e) Bs[(cc * 8 + e) * 64 + n] = pk.s[e];
        }
        __syncthreads();
        v16bf af0 = frag_a(&As[(wm * 32) * 32], 32);
        v16bf af1 = frag_a(&As[(wm * 32 + 16) * 32], 32);
        v16bf bf0 = frag_b(&Bs[wn * 32], 64);
        v16bf bf1 = frag_b(&Bs[wn * 32 + 16], 64);
        acc[0][0] = __builtin_amdgcn_wmma_f32_16x16x32_bf16(false, af0, false, bf0, (short)0, acc[0][0], false, false);
        acc[0][1] = __builtin_amdgcn_wmma_f32_16x16x32_bf16(false, af0, false, bf1, (short)0, acc[0][1], false, false);
        acc[1][0] = __builtin_amdgcn_wmma_f32_16x16x32_bf16(false, af1, false, bf0, (short)0, acc[1][0], false, false);
        acc[1][1] = __builtin_amdgcn_wmma_f32_16x16x32_bf16(false, af1, false, bf1, (short)0, acc[1][1], false, false);
        __syncthreads();
    }

    // epilogue (C/D layout: lane n = L&15, row = vgpr + 8*(L>>4))
    const int half = lane >> 4, l15 = lane & 15;
    #pragma unroll
    for (int i = 0; i < 2; ++i)
        #pragma unroll
        for (int j = 0; j < 2; ++j)
            #pragma unroll
            for (int r = 0; r < 8; ++r) {
                int row = rowBase + wm * 32 + 16 * i + r + 8 * half;
                int col = colBase + wn * 32 + 16 * j + l15;
                if (row < M && col < Nout) {
                    float v = acc[i][j][r] + bias[col];
                    if (RESID) v += resid[(size_t)row * Nout + col];
                    if (GELU)  v = 0.5f * v * (1.0f + erff(v * 0.70710678118654752f));
                    if (O32) out32[(size_t)row * Nout + col] = v;
                    if (O16) out16[(size_t)row * Nout + col] = f2bf(v);
                }
            }
}

// ---------------- fused attention per (b, h) ---------------------------------
// K^T and V staged fully in LDS (CDNA5: 320KB/WGP); per-wave 16-row Q blocks,
// S = Q K^T via WMMA, softmax in registers (shfl within 16-lane halves),
// P bounced through per-wave LDS tile to re-layout C->A, then O = P V via WMMA.
__global__ __launch_bounds__(256)
void attn_kernel(const ushort_t* __restrict__ qkv, ushort_t* __restrict__ out) {
    extern __shared__ ushort_t sm[];
    ushort_t* Kt = sm;                   // [64][224]  d-major K^T
    ushort_t* V  = sm + 64 * 224;        // [224][64]  row-major V (zero-padded)
    ushort_t* P  = sm + 2 * 64 * 224;    // [8 waves][16][224]
    const int bh = blockIdx.x;
    const int b = bh / NHEAD, hh = bh % NHEAD;
    const int tid = threadIdx.x, lane = tid & 31, wave = tid >> 5;
    const int half = lane >> 4, l15 = lane & 15;
    const size_t rowPitch = 3 * DMODEL;  // 2304
    const size_t qkvBase = (size_t)b * NTOK * rowPitch;

    for (int idx = tid; idx < 224 * 64; idx += 256) {
        int j = idx >> 6, d = idx & 63;
        ushort_t kv = 0, vv = 0;
        if (j < NTOK) {
            const ushort_t* row = qkv + qkvBase + (size_t)j * rowPitch + hh * HDIM + d;
            kv = row[DMODEL];        // K block at col offset 768
            vv = row[2 * DMODEL];    // V block at col offset 1536
        }
        Kt[d * 224 + j] = kv;
        V[idx] = vv;
    }
    __syncthreads();

    ushort_t* Pw = P + wave * 16 * 224;

    for (int qb = wave; qb < 13; qb += 8) {
        const int row0 = qb * 16;
        // Q fragments directly from global (rows clamped; garbage rows unused)
        v16bf qf[2];
        #pragma unroll
        for (int kk = 0; kk < 2; ++kk) {
            int trow = row0 + l15; if (trow > NTOK - 1) trow = NTOK - 1;
            const ushort_t* r = qkv + qkvBase + (size_t)trow * rowPitch + hh * HDIM + kk * 32 + 8 * half;
            FragU f;
            f.u[0] = *(const u32x4*)(r);
            f.u[1] = *(const u32x4*)(r + 16);
            qf[kk] = f.v;
        }
        // S = Q K^T  (16 x 224 in 14 accumulators)
        v8f sacc[14];
        #pragma unroll
        for (int t = 0; t < 14; ++t) sacc[t] = (v8f)0.f;
        #pragma unroll
        for (int kk = 0; kk < 2; ++kk)
            #pragma unroll
            for (int t = 0; t < 14; ++t) {
                v16bf kf = frag_b(&Kt[(kk * 32) * 224 + 16 * t], 224);
                sacc[t] = __builtin_amdgcn_wmma_f32_16x16x32_bf16(false, qf[kk], false, kf, (short)0, sacc[t], false, false);
            }
        // softmax over 224 cols (mask j >= 197)
        float rmax[8];
        #pragma unroll
        for (int r = 0; r < 8; ++r) rmax[r] = -3.0e38f;
        #pragma unroll
        for (int t = 0; t < 14; ++t) {
            int j = 16 * t + l15;
            #pragma unroll
            for (int r = 0; r < 8; ++r) {
                float s = sacc[t][r] * ATTN_SCALE;
                if (j >= NTOK) s = -3.0e38f;
                sacc[t][r] = s;
                rmax[r] = fmaxf(rmax[r], s);
            }
        }
        #pragma unroll
        for (int r = 0; r < 8; ++r)
            #pragma unroll
            for (int off = 1; off < 16; off <<= 1)
                rmax[r] = fmaxf(rmax[r], __shfl_xor(rmax[r], off, 32));
        float rsum[8];
        #pragma unroll
        for (int r = 0; r < 8; ++r) rsum[r] = 0.f;
        #pragma unroll
        for (int t = 0; t < 14; ++t)
            #pragma unroll
            for (int r = 0; r < 8; ++r) {
                float p = expf(sacc[t][r] - rmax[r]);
                sacc[t][r] = p;
                rsum[r] += p;
            }
        #pragma unroll
        for (int r = 0; r < 8; ++r)
            #pragma unroll
            for (int off = 1; off < 16; off <<= 1)
                rsum[r] += __shfl_xor(rsum[r], off, 32);
        // write normalized P (bf16) to this wave's LDS tile (C-layout scatter)
        #pragma unroll
        for (int r = 0; r < 8; ++r) {
            float inv = 1.0f / rsum[r];
            #pragma unroll
            for (int t = 0; t < 14; ++t)
                Pw[(r + 8 * half) * 224 + 16 * t + l15] = f2bf(sacc[t][r] * inv);
        }
        // O = P V  (16 x 64)
        v8f oacc[4];
        #pragma unroll
        for (int nt = 0; nt < 4; ++nt) oacc[nt] = (v8f)0.f;
        #pragma unroll
        for (int jt = 0; jt < 7; ++jt) {
            v16bf pf = frag_a(&Pw[jt * 32], 224);
            #pragma unroll
            for (int nt = 0; nt < 4; ++nt) {
                v16bf vf = frag_b(&V[(jt * 32) * 64 + 16 * nt], 64);
                oacc[nt] = __builtin_amdgcn_wmma_f32_16x16x32_bf16(false, pf, false, vf, (short)0, oacc[nt], false, false);
            }
        }
        #pragma unroll
        for (int nt = 0; nt < 4; ++nt)
            #pragma unroll
            for (int r = 0; r < 8; ++r) {
                int tok = row0 + r + 8 * half;
                if (tok < NTOK)
                    out[((size_t)(b * NTOK + tok)) * DMODEL + hh * HDIM + 16 * nt + l15] = f2bf(oacc[nt][r]);
            }
    }
}

// ---------------- host orchestration ----------------------------------------
extern "C" void kernel_launch(void* const* d_in, const int* in_sizes, int n_in,
                              void* d_out, int out_size, void* d_ws, size_t ws_size,
                              hipStream_t stream) {
    (void)in_sizes; (void)n_in; (void)out_size; (void)ws_size;
    const float* x      = (const float*)d_in[0];
    const float* conv_w = (const float*)d_in[1];
    const float* conv_b = (const float*)d_in[2];
    const float* cls    = (const float*)d_in[3];
    const float* pos    = (const float*)d_in[4];
    const float* qkv_w  = (const float*)d_in[5];
    const float* qkv_b  = (const float*)d_in[6];
    const float* proj_w = (const float*)d_in[7];
    const float* proj_b = (const float*)d_in[8];
    const float* n1_s   = (const float*)d_in[9];
    const float* n1_b   = (const float*)d_in[10];
    const float* n2_s   = (const float*)d_in[11];
    const float* n2_b   = (const float*)d_in[12];
    const float* fc1_w  = (const float*)d_in[13];
    const float* fc1_b  = (const float*)d_in[14];
    const float* fc2_w  = (const float*)d_in[15];
    const float* fc2_b  = (const float*)d_in[16];
    const float* enc_s  = (const float*)d_in[17];
    const float* enc_b  = (const float*)d_in[18];
    const float* head_w = (const float*)d_in[19];
    const float* head_b = (const float*)d_in[20];
    float* outp = (float*)d_out;

    char* ws = (char*)d_ws;
    size_t cur = 0;
    auto alloc = [&](size_t bytes) -> char* {
        char* p = ws + cur;
        cur = (cur + bytes + 255) & ~(size_t)255;
        return p;
    };
    const size_t E_QKVW = (size_t)NDEPTH * 3 * DMODEL * DMODEL;
    const size_t E_PRJW = (size_t)NDEPTH * DMODEL * DMODEL;
    const size_t E_FC1W = (size_t)NDEPTH * DMLP * DMODEL;
    const size_t E_FC2W = (size_t)NDEPTH * DMODEL * DMLP;

    ushort_t* wc    = (ushort_t*)alloc((size_t)DMODEL * DMODEL * 2);
    ushort_t* wqkv  = (ushort_t*)alloc(E_QKVW * 2);
    ushort_t* wproj = (ushort_t*)alloc(E_PRJW * 2);
    ushort_t* wfc1  = (ushort_t*)alloc(E_FC1W * 2);
    ushort_t* wfc2  = (ushort_t*)alloc(E_FC2W * 2);
    ushort_t* whead = (ushort_t*)alloc((size_t)NCLS * DMODEL * 2);
    ushort_t* Xp    = (ushort_t*)alloc((size_t)MPATCH * DMODEL * 2);
    float*    P32   = (float*)   alloc((size_t)MPATCH * DMODEL * 4);
    float*    hbuf  = (float*)   alloc((size_t)MROWS * DMODEL * 4);
    ushort_t* xln   = (ushort_t*)alloc((size_t)MROWS * DMODEL * 2);
    ushort_t* qkvb  = (ushort_t*)alloc((size_t)MROWS * 3 * DMODEL * 2);
    ushort_t* attnb = (ushort_t*)alloc((size_t)MROWS * DMODEL * 2);
    ushort_t* mlpb  = (ushort_t*)alloc((size_t)MROWS * DMLP * 2);
    ushort_t* clsln = (ushort_t*)alloc((size_t)BATCH * DMODEL * 2);

    auto cvt = [&](const float* src, ushort_t* dst, long n) {
        unsigned blocks = (unsigned)((n + 255) / 256);
        cvt_kernel<<<dim3(blocks), dim3(256), 0, stream>>>(src, dst, n);
    };
    // one-time (per call) weight downconversion: bf16 weights (~173MB) fit in L2
    cvt(conv_w, wc,    (long)DMODEL * DMODEL);
    cvt(qkv_w,  wqkv,  (long)E_QKVW);
    cvt(proj_w, wproj, (long)E_PRJW);
    cvt(fc1_w,  wfc1,  (long)E_FC1W);
    cvt(fc2_w,  wfc2,  (long)E_FC2W);
    cvt(head_w, whead, (long)NCLS * DMODEL);

    // patch embed
    {
        long n = (long)MPATCH * DMODEL;
        im2col_kernel<<<dim3((unsigned)((n + 255) / 256)), dim3(256), 0, stream>>>(x, Xp);
        dim3 grid((DMODEL + 63) / 64, (MPATCH + 127) / 128);
        gemm_bt_kernel<false, false, true, false><<<grid, dim3(256), 0, stream>>>(
            Xp, DMODEL, wc, DMODEL, conv_b, nullptr, P32, nullptr, MPATCH, DMODEL, DMODEL);
        long n2 = (long)MROWS * DMODEL;
        assemble_kernel<<<dim3((unsigned)((n2 + 255) / 256)), dim3(256), 0, stream>>>(P32, cls, pos, hbuf);
    }

    static int smemSet = 0; // idempotent attribute set (deterministic either way)
    (void)smemSet;
    hipFuncSetAttribute((const void*)attn_kernel, hipFuncAttributeMaxDynamicSharedMemorySize, 114688);

    for (int L = 0; L < NDEPTH; ++L) {
        const ushort_t* wq = wqkv  + (size_t)L * 3 * DMODEL * DMODEL;
        const ushort_t* wp = wproj + (size_t)L * DMODEL * DMODEL;
        const ushort_t* w1 = wfc1  + (size_t)L * DMLP * DMODEL;
        const ushort_t* w2 = wfc2  + (size_t)L * DMODEL * DMLP;

        ln_kernel<<<dim3(MROWS), dim3(256), 0, stream>>>(hbuf, DMODEL, n1_s + (size_t)L * DMODEL,
                                                         n1_b + (size_t)L * DMODEL, xln, MROWS);
        {
            dim3 grid((3 * DMODEL + 63) / 64, (MROWS + 127) / 128);
            gemm_bt_kernel<false, false, false, true><<<grid, dim3(256), 0, stream>>>(
                xln, DMODEL, wq, DMODEL, qkv_b + (size_t)L * 3 * DMODEL, nullptr,
                nullptr, qkvb, MROWS, 3 * DMODEL, DMODEL);
        }
        attn_kernel<<<dim3(BATCH * NHEAD), dim3(256), 114688, stream>>>(qkvb, attnb);
        {
            dim3 grid((DMODEL + 63) / 64, (MROWS + 127) / 128);
            gemm_bt_kernel<false, true, true, false><<<grid, dim3(256), 0, stream>>>(
                attnb, DMODEL, wp, DMODEL, proj_b + (size_t)L * DMODEL, hbuf,
                hbuf, nullptr, MROWS, DMODEL, DMODEL);
        }
        ln_kernel<<<dim3(MROWS), dim3(256), 0, stream>>>(hbuf, DMODEL, n2_s + (size_t)L * DMODEL,
                                                         n2_b + (size_t)L * DMODEL, xln, MROWS);
        {
            dim3 grid((DMLP + 63) / 64, (MROWS + 127) / 128);
            gemm_bt_kernel<true, false, false, true><<<grid, dim3(256), 0, stream>>>(
                xln, DMODEL, w1, DMODEL, fc1_b + (size_t)L * DMLP, nullptr,
                nullptr, mlpb, MROWS, DMLP, DMODEL);
        }
        {
            dim3 grid((DMODEL + 63) / 64, (MROWS + 127) / 128);
            gemm_bt_kernel<false, true, true, false><<<grid, dim3(256), 0, stream>>>(
                mlpb, DMLP, w2, DMLP, fc2_b + (size_t)L * DMODEL, hbuf,
                hbuf, nullptr, MROWS, DMODEL, DMLP);
        }
    }

    // final LN only on cls rows, then head
    ln_kernel<<<dim3(BATCH), dim3(256), 0, stream>>>(hbuf, (long)NTOK * DMODEL, enc_s, enc_b, clsln, BATCH);
    {
        dim3 grid((NCLS + 63) / 64, (BATCH + 127) / 128);
        gemm_bt_kernel<false, false, true, false><<<grid, dim3(256), 0, stream>>>(
            clsln, DMODEL, whead, DMODEL, head_b, nullptr, outp, nullptr, BATCH, NCLS, DMODEL);
    }
}
